// AGCRNCell_3616362463697
// MI455X (gfx1250) — compile-verified, matrix operand order checked by reading, not run on hardware
//
#include <hip/hip_runtime.h>

#define B_DIM 64
#define N_NODES 2048
#define D_EMB 16
#define C_IN 66
#define COLS (B_DIM * C_IN)   /* 4224 */
#define HID 64
#define OG 128                /* gate output = 2*HID */
#define KPAD 160              /* padded reduction dim for node GEMM: 2 sections of 80 */
#define LDJ 168               /* LDS row stride in halves (84 dwords -> conflict-free) */
#define GLDK 40               /* GEMM LDS row stride in halves (20 dwords -> conflict-free) */

typedef __bf16 bf16_t;
typedef __attribute__((ext_vector_type(16))) __bf16 v16bf;
typedef __attribute__((ext_vector_type(8)))  float  v8f;

union FragU { v16bf v; uint4 q[2]; };

__device__ inline bf16_t tob(float f) { return (bf16_t)f; }   // hw cvt, RNE
__device__ inline float  tof(bf16_t h) { return (float)h; }

// ISA 16-bit A/B fragment layout (16x32 / 32x16): lane group (lane>>4) selects
// K offset 0 or 8; VGPRs 0-3 = K ksel..ksel+7, VGPRs 4-7 = K ksel+16..ksel+23.
__device__ inline v16bf load_frag(const bf16_t* p) {
  FragU f;
  f.q[0] = *(const uint4*)(p);
  f.q[1] = *(const uint4*)(p + 16);
  return f.v;
}

__device__ inline v8f wmma_bf16(v16bf a, v16bf b, v8f c) {
  return __builtin_amdgcn_wmma_f32_16x16x32_bf16(false, a, false, b, (short)0, c,
                                                 false, false);
}

// CDNA5 async copy global->LDS (tracked by ASYNCcnt). lds_off is the LDS byte
// offset (low 32 bits of the generic pointer to a __shared__ object).
__device__ inline void async_ld_b128(unsigned lds_off, const bf16_t* g) {
  asm volatile("global_load_async_to_lds_b128 %0, %1, off"
               :: "v"(lds_off), "v"(g)
               : "memory");
}

// ---------------------------------------------------------------------------
// f32 -> bf16 conversion (one-time weight repack to halve L2 traffic)
// ---------------------------------------------------------------------------
__global__ __launch_bounds__(256) void cvt_bf16(
    const float* __restrict__ src, bf16_t* __restrict__ dst, int n) {
  int i = blockIdx.x * 256 + threadIdx.x;
  if (i < n) dst[i] = tob(src[i]);
}

// ---------------------------------------------------------------------------
// supports[n][m] = softmax_m(relu(E[n] . E[m]))  -> bf16
// ---------------------------------------------------------------------------
__global__ __launch_bounds__(256) void supports_kernel(
    const float* __restrict__ E, bf16_t* __restrict__ S) {
  const int n = blockIdx.x, tid = threadIdx.x;
  const int lane = tid & 31, wave = tid >> 5;
  __shared__ float eN[D_EMB];
  __shared__ float red[8];
  if (tid < D_EMB) eN[tid] = E[n * D_EMB + tid];
  __syncthreads();

  float vals[8];
  float mx = 0.f;
  const int mbase = tid * 8;
#pragma unroll
  for (int i = 0; i < 8; i++) {
    const float* em = E + (size_t)(mbase + i) * D_EMB;
    float dot = 0.f;
#pragma unroll
    for (int d = 0; d < D_EMB; d++) dot += eN[d] * em[d];
    float v = dot > 0.f ? dot : 0.f;
    vals[i] = v;
    mx = fmaxf(mx, v);
  }
  for (int off = 16; off >= 1; off >>= 1) mx = fmaxf(mx, __shfl_xor(mx, off));
  if (lane == 0) red[wave] = mx;
  __syncthreads();
  float bm = red[0];
#pragma unroll
  for (int i = 1; i < 8; i++) bm = fmaxf(bm, red[i]);

  float sum = 0.f;
#pragma unroll
  for (int i = 0; i < 8; i++) { vals[i] = __expf(vals[i] - bm); sum += vals[i]; }
  for (int off = 16; off >= 1; off >>= 1) sum += __shfl_xor(sum, off);
  __syncthreads();
  if (lane == 0) red[wave] = sum;
  __syncthreads();
  float ts = 0.f;
#pragma unroll
  for (int i = 0; i < 8; i++) ts += red[i];
  const float inv = 1.f / ts;
#pragma unroll
  for (int i = 0; i < 8; i++)
    S[(size_t)n * N_NODES + mbase + i] = tob(vals[i] * inv);
}

// ---------------------------------------------------------------------------
// pack concat(x, state) -> bf16 in two layouts:
//   cm[col][m]  (K-contiguous for GEMM B fragments), rm[n][b*66+c]
// ---------------------------------------------------------------------------
__global__ __launch_bounds__(256) void pack_gate(
    const float* __restrict__ x, const float* __restrict__ state,
    bf16_t* __restrict__ cm, bf16_t* __restrict__ rm) {
  size_t idx = (size_t)blockIdx.x * 256 + threadIdx.x;
  if (idx >= (size_t)B_DIM * N_NODES * C_IN) return;
  int c = (int)(idx % C_IN);
  size_t bn = idx / C_IN;             // b*N + n
  int nn = (int)(bn % N_NODES);
  int b = (int)(bn / N_NODES);
  float v = (c < 2) ? x[bn * 2 + c] : state[bn * HID + (c - 2)];
  bf16_t h = tob(v);
  rm[(size_t)nn * COLS + b * C_IN + c] = h;
  cm[((size_t)(b * C_IN + c)) * N_NODES + nn] = h;
}

__global__ __launch_bounds__(256) void pack_cand(
    const float* __restrict__ x, const float* __restrict__ state,
    const bf16_t* __restrict__ zr,
    bf16_t* __restrict__ cm, bf16_t* __restrict__ rm) {
  size_t idx = (size_t)blockIdx.x * 256 + threadIdx.x;
  if (idx >= (size_t)B_DIM * N_NODES * C_IN) return;
  int c = (int)(idx % C_IN);
  size_t bn = idx / C_IN;
  int nn = (int)(bn % N_NODES);
  int b = (int)(bn / N_NODES);
  float v;
  if (c < 2) {
    v = x[bn * 2 + c];
  } else {
    float r = tof(zr[((size_t)nn * B_DIM + b) * OG + HID + (c - 2)]);
    v = r * state[bn * HID + (c - 2)];
  }
  bf16_t h = tob(v);
  rm[(size_t)nn * COLS + b * C_IN + c] = h;
  cm[((size_t)(b * C_IN + c)) * N_NODES + nn] = h;
}

// ---------------------------------------------------------------------------
// Sx[n][col] = sum_m supports[n][m] * inp[col][m]   (2048 x 4224 x 2048)
// Block tile 128x128, 8 waves (4M x 2N), wave tile 32x64 (8 WMMA / K-step).
// A/B tiles triple-buffered in LDS via async global->LDS copies; one barrier
// per K-step: [issue -> buf[p+1]; s_wait_asynccnt<=4; barrier; read buf[p]].
// ---------------------------------------------------------------------------
__global__ __launch_bounds__(256) void gemm_sx(
    const bf16_t* __restrict__ A,   // [2048][2048] supports
    const bf16_t* __restrict__ Bt,  // [4224][2048] (B^T, K-contiguous)
    bf16_t* __restrict__ Cm) {      // [2048][4224]
  __shared__ bf16_t As[3][128 * GLDK];
  __shared__ bf16_t Bs[3][128 * GLDK];

  const int tid = threadIdx.x;
  const int lane = tid & 31, wave = tid >> 5;
  const int m0 = blockIdx.x * 128;
  const int n0 = blockIdx.y * 128;

  // cooperative staging: thread -> (row pair, 16B chunk of the 64B k-row)
  const int ch = tid & 3;        // chunk: k offset = ch*8 halves
  const int rowc = tid >> 2;     // 0..63 (rows rowc and rowc+64)
  const unsigned asb = (unsigned)(uintptr_t)&As[0][0];
  const unsigned bsb = (unsigned)(uintptr_t)&Bs[0][0];
  const unsigned bufBytes = 128 * GLDK * 2;

  const int rowl = lane & 15, ksel = (lane >> 4) * 8, hi = lane >> 4;
  const int wm = wave & 3;       // M subtile (32 rows)
  const int wn = wave >> 2;      // N subtile (64 cols)

  v8f acc[2][4] = {};

  // prologue: stage k=0 into buffer 0
#pragma unroll
  for (int i = 0; i < 2; i++) {
    int row = rowc + i * 64;
    unsigned loff = (unsigned)((row * GLDK + ch * 8) * 2);
    async_ld_b128(asb + loff, A + (size_t)(m0 + row) * N_NODES + ch * 8);
    async_ld_b128(bsb + loff, Bt + (size_t)(n0 + row) * N_NODES + ch * 8);
  }

  int p = 0;
  for (int k = 0; k < N_NODES; k += 32) {
    // stage next K-step into buffer p+1 (last iter: harmless reload).
    // Its previous readers were 2 barriers back -> WAR-safe.
    int kn = (k + 32 < N_NODES) ? (k + 32) : k;
    int pn = (p == 2) ? 0 : p + 1;
    unsigned pb = (unsigned)(pn * bufBytes);
#pragma unroll
    for (int i = 0; i < 2; i++) {
      int row = rowc + i * 64;
      unsigned loff = pb + (unsigned)((row * GLDK + ch * 8) * 2);
      async_ld_b128(asb + loff, A + (size_t)(m0 + row) * N_NODES + kn + ch * 8);
      async_ld_b128(bsb + loff, Bt + (size_t)(n0 + row) * N_NODES + kn + ch * 8);
    }
    // async loads complete in order: <=4 outstanding => current tile resident
    asm volatile("s_wait_asynccnt 0x4" ::: "memory");
    __syncthreads();

    const bf16_t* aBase = &As[p][0];
    const bf16_t* bBase = &Bs[p][0];
    v16bf bf[4];
#pragma unroll
    for (int t = 0; t < 4; t++)
      bf[t] = load_frag(bBase + (wn * 64 + t * 16 + rowl) * GLDK + ksel);
    v16bf a0 = load_frag(aBase + (wm * 32 + rowl) * GLDK + ksel);
    v16bf a1 = load_frag(aBase + (wm * 32 + 16 + rowl) * GLDK + ksel);
#pragma unroll
    for (int t = 0; t < 4; t++) {
      acc[0][t] = wmma_bf16(a0, bf[t], acc[0][t]);
      acc[1][t] = wmma_bf16(a1, bf[t], acc[1][t]);
    }
    p = pn;
  }

#pragma unroll
  for (int mt = 0; mt < 2; mt++)
#pragma unroll
    for (int t = 0; t < 4; t++)
#pragma unroll
      for (int r = 0; r < 8; r++) {
        int m = m0 + wm * 32 + mt * 16 + r + hi * 8;
        int col = n0 + wn * 64 + t * 16 + rowl;
        Cm[(size_t)m * COLS + col] = tob(acc[mt][t][r]);
      }
}

// ---------------------------------------------------------------------------
// Per-node gate: G[b,o] = sum_j A[b,j]*W_n[j,o] + bias; z_r = sigmoid(G)
// A (64x160) and W_n^T (128x160) staged in LDS (bf16, stride 168 halves).
// ---------------------------------------------------------------------------
__global__ __launch_bounds__(256) void node_gate(
    const float* __restrict__ E, const bf16_t* __restrict__ gw,
    const float* __restrict__ gb,
    const bf16_t* __restrict__ inp_rm,
    const bf16_t* __restrict__ sx_rm,
    bf16_t* __restrict__ zr) {
  const int n = blockIdx.x, tid = threadIdx.x;
  __shared__ float eN[D_EMB];
  __shared__ float bias[OG];
  __shared__ bf16_t Alds[B_DIM * LDJ];
  __shared__ bf16_t Wlds[OG * LDJ];

  if (tid < D_EMB) eN[tid] = E[n * D_EMB + tid];
  __syncthreads();

  for (int idx = tid; idx < B_DIM * KPAD; idx += 256) {
    int b = idx / KPAD, j = idx % KPAD;
    int k = j / 80, cc = j % 80;
    bf16_t v = tob(0.f);
    if (cc < C_IN) {
      const bf16_t* src = (k == 0) ? inp_rm : sx_rm;
      v = src[(size_t)n * COLS + b * C_IN + cc];
    }
    Alds[b * LDJ + j] = v;
  }
  // o-fastest mapping: consecutive lanes read consecutive o -> coalesced
  for (int idx = tid; idx < OG * KPAD; idx += 256) {
    int o = idx & (OG - 1), j = idx >> 7;
    int k = j / 80, cc = j % 80;
    float s = 0.f;
    if (cc < C_IN) {
#pragma unroll
      for (int d = 0; d < D_EMB; d++)
        s += eN[d] * tof(gw[((size_t)(d * 2 + k) * C_IN + cc) * OG + o]);
    }
    Wlds[o * LDJ + j] = tob(s);
  }
  for (int o = tid; o < OG; o += 256) {
    float s = 0.f;
#pragma unroll
    for (int d = 0; d < D_EMB; d++) s += eN[d] * gb[d * OG + o];
    bias[o] = s;
  }
  __syncthreads();

  const int lane = tid & 31, wave = tid >> 5;
  const int rowl = lane & 15, ksel = (lane >> 4) * 8, hi = lane >> 4;
  const int mtile = wave >> 1;
  const int nb = (wave & 1) * 4;
  v8f acc[4] = {};
  const bf16_t* aPtr = Alds + (mtile * 16 + rowl) * LDJ + ksel;
  for (int jb = 0; jb < KPAD; jb += 32) {
    v16bf a = load_frag(aPtr + jb);
#pragma unroll
    for (int t = 0; t < 4; t++) {
      v16bf w = load_frag(Wlds + ((nb + t) * 16 + rowl) * LDJ + ksel + jb);
      acc[t] = wmma_bf16(a, w, acc[t]);
    }
  }
#pragma unroll
  for (int t = 0; t < 4; t++)
#pragma unroll
    for (int r = 0; r < 8; r++) {
      int b = mtile * 16 + r + hi * 8;
      int o = (nb + t) * 16 + rowl;
      float v = acc[t][r] + bias[o];
      v = 1.f / (1.f + __expf(-v));
      zr[((size_t)n * B_DIM + b) * OG + o] = tob(v);
    }
}

// ---------------------------------------------------------------------------
// Per-node update: hc = tanh(G + bias); out = z*state + (1-z)*hc
// ---------------------------------------------------------------------------
__global__ __launch_bounds__(256) void node_update(
    const float* __restrict__ E, const bf16_t* __restrict__ uw,
    const float* __restrict__ ub,
    const bf16_t* __restrict__ inp_rm,
    const bf16_t* __restrict__ sx_rm,
    const bf16_t* __restrict__ zr,
    const float* __restrict__ state, float* __restrict__ out) {
  const int n = blockIdx.x, tid = threadIdx.x;
  __shared__ float eN[D_EMB];
  __shared__ float bias[HID];
  __shared__ bf16_t Alds[B_DIM * LDJ];
  __shared__ bf16_t Wlds[HID * LDJ];

  if (tid < D_EMB) eN[tid] = E[n * D_EMB + tid];
  __syncthreads();

  for (int idx = tid; idx < B_DIM * KPAD; idx += 256) {
    int b = idx / KPAD, j = idx % KPAD;
    int k = j / 80, cc = j % 80;
    bf16_t v = tob(0.f);
    if (cc < C_IN) {
      const bf16_t* src = (k == 0) ? inp_rm : sx_rm;
      v = src[(size_t)n * COLS + b * C_IN + cc];
    }
    Alds[b * LDJ + j] = v;
  }
  for (int idx = tid; idx < HID * KPAD; idx += 256) {
    int o = idx & (HID - 1), j = idx >> 6;
    int k = j / 80, cc = j % 80;
    float s = 0.f;
    if (cc < C_IN) {
#pragma unroll
      for (int d = 0; d < D_EMB; d++)
        s += eN[d] * tof(uw[((size_t)(d * 2 + k) * C_IN + cc) * HID + o]);
    }
    Wlds[o * LDJ + j] = tob(s);
  }
  for (int o = tid; o < HID; o += 256) {
    float s = 0.f;
#pragma unroll
    for (int d = 0; d < D_EMB; d++) s += eN[d] * ub[d * HID + o];
    bias[o] = s;
  }
  __syncthreads();

  const int lane = tid & 31, wave = tid >> 5;
  const int rowl = lane & 15, ksel = (lane >> 4) * 8, hi = lane >> 4;
  const int mtile = wave >> 1;
  const int nb = (wave & 1) * 2;
  v8f acc[2] = {};
  const bf16_t* aPtr = Alds + (mtile * 16 + rowl) * LDJ + ksel;
  for (int jb = 0; jb < KPAD; jb += 32) {
    v16bf a = load_frag(aPtr + jb);
#pragma unroll
    for (int t = 0; t < 2; t++) {
      v16bf w = load_frag(Wlds + ((nb + t) * 16 + rowl) * LDJ + ksel + jb);
      acc[t] = wmma_bf16(a, w, acc[t]);
    }
  }
#pragma unroll
  for (int t = 0; t < 2; t++)
#pragma unroll
    for (int r = 0; r < 8; r++) {
      int b = mtile * 16 + r + hi * 8;
      int o = (nb + t) * 16 + rowl;
      float hc = tanhf(acc[t][r] + bias[o]);
      float z = tof(zr[((size_t)n * B_DIM + b) * OG + o]);
      float st = state[((size_t)b * N_NODES + n) * HID + o];
      out[((size_t)b * N_NODES + n) * HID + o] = z * st + (1.f - z) * hc;
    }
}

// ---------------------------------------------------------------------------
extern "C" void kernel_launch(void* const* d_in, const int* in_sizes, int n_in,
                              void* d_out, int out_size, void* d_ws,
                              size_t ws_size, hipStream_t stream) {
  const float* x = (const float*)d_in[0];
  const float* state = (const float*)d_in[1];
  const float* E = (const float*)d_in[2];
  const float* gate_w = (const float*)d_in[3];
  const float* gate_b = (const float*)d_in[4];
  const float* update_w = (const float*)d_in[5];
  const float* update_b = (const float*)d_in[6];
  float* out = (float*)d_out;

  char* ws = (char*)d_ws;
  bf16_t* supports = (bf16_t*)ws; ws += (size_t)N_NODES * N_NODES * 2;
  bf16_t* inp_cm = (bf16_t*)ws;   ws += (size_t)COLS * N_NODES * 2;
  bf16_t* inp_rm = (bf16_t*)ws;   ws += (size_t)N_NODES * COLS * 2;
  bf16_t* sx_rm  = (bf16_t*)ws;   ws += (size_t)N_NODES * COLS * 2;
  bf16_t* zr     = (bf16_t*)ws;   ws += (size_t)N_NODES * B_DIM * OG * 2;
  bf16_t* gwb    = (bf16_t*)ws;   ws += (size_t)D_EMB * 2 * C_IN * OG * 2;
  bf16_t* uwb    = (bf16_t*)ws;

  const int nGW = D_EMB * 2 * C_IN * OG;   // 270336
  const int nUW = D_EMB * 2 * C_IN * HID;  // 135168
  const size_t packN = (size_t)B_DIM * N_NODES * C_IN;
  const int packBlocks = (int)((packN + 255) / 256);
  const dim3 gemmGrid(N_NODES / 128, COLS / 128);

  cvt_bf16<<<(nGW + 255) / 256, 256, 0, stream>>>(gate_w, gwb, nGW);
  cvt_bf16<<<(nUW + 255) / 256, 256, 0, stream>>>(update_w, uwb, nUW);
  supports_kernel<<<N_NODES, 256, 0, stream>>>(E, supports);
  pack_gate<<<packBlocks, 256, 0, stream>>>(x, state, inp_cm, inp_rm);
  gemm_sx<<<gemmGrid, 256, 0, stream>>>(supports, inp_cm, sx_rm);
  node_gate<<<N_NODES, 256, 0, stream>>>(E, gwb, gate_b, inp_rm, sx_rm, zr);
  pack_cand<<<packBlocks, 256, 0, stream>>>(x, state, zr, inp_cm, inp_rm);
  gemm_sx<<<gemmGrid, 256, 0, stream>>>(supports, inp_cm, sx_rm);
  node_update<<<N_NODES, 256, 0, stream>>>(E, uwb, update_b, inp_rm, sx_rm,
                                           zr, state, out);
}
